// Block_25752623907165
// MI455X (gfx1250) — matrix operand order repaired and checked by reading, not compiled, not run on hardware
//
#include <hip/hip_runtime.h>

// ---------------------------------------------------------------------------
// Types
// ---------------------------------------------------------------------------
typedef __bf16 bf16;
typedef __attribute__((ext_vector_type(16))) bf16 bf16x16;
typedef __attribute__((ext_vector_type(8)))  float f32x8;

struct __align__(16) U4 { unsigned int x, y, z, w; };
union AF { bf16x16 v; U4 u[2]; };           // 32B WMMA 16-bit operand fragment
union HC { bf16 h; unsigned short s; };     // bf16 <-> bits

static __device__ __forceinline__
f32x8 wmma_bf16(const AF& a, const AF& b, f32x8 c) {
    return __builtin_amdgcn_wmma_f32_16x16x32_bf16(
        /*neg_a=*/false, a.v, /*neg_b=*/false, b.v,
        /*c_mod=*/(short)0, c, /*reuse_a=*/false, /*reuse_b=*/false);
}

// Low 32 bits of a generic pointer into the shared aperture = LDS byte address
// (ISA 10.2).
static __device__ __forceinline__ unsigned ldsoff(const void* p) {
    return (unsigned)(unsigned long long)p;
}

// Direct global -> LDS async copy, 16B per lane, tracked by ASYNCcnt
// (CDNA5 ISA op 98 / CDNA5 ISA).
static __device__ __forceinline__
void async_copy_b128(unsigned lds, const void* g) {
    asm volatile("global_load_async_to_lds_b128 %0, %1, off"
                 :: "v"(lds), "v"((unsigned long long)g)
                 : "memory");
}
static __device__ __forceinline__ void wait_async0() {
    asm volatile("s_wait_asynccnt 0x0" ::: "memory");
}

#define MROWS 8192      // B*N = 8*1024 token rows
#define CEMB  1024

// ---------------------------------------------------------------------------
// Weight convert + transpose: fp32 [K,N] -> bf16 [N,K]
// ---------------------------------------------------------------------------
__global__ __launch_bounds__(256)
void wt_transpose(const float* __restrict__ in, bf16* __restrict__ outT,
                  int K, int N) {
    __shared__ float tile[32][33];
    int n0 = blockIdx.x * 32, k0 = blockIdx.y * 32;
    int tx = threadIdx.x, ty = threadIdx.y;   // 32 x 8
#pragma unroll
    for (int i = 0; i < 4; ++i)
        tile[ty * 4 + i][tx] = in[(size_t)(k0 + ty * 4 + i) * N + n0 + tx];
    __syncthreads();
#pragma unroll
    for (int i = 0; i < 4; ++i)
        outT[(size_t)(n0 + ty * 4 + i) * K + k0 + tx] = (bf16)tile[tx][ty * 4 + i];
}

// ---------------------------------------------------------------------------
// LayerNorm: fp32 [rows,1024] -> bf16 normalized
// ---------------------------------------------------------------------------
__global__ __launch_bounds__(256)
void ln_kernel(const float* __restrict__ x, const float* __restrict__ g,
               const float* __restrict__ b, bf16* __restrict__ out) {
    int row = blockIdx.x, t = threadIdx.x;
    const float* xr = x + (size_t)row * CEMB;
    float vals[4], s = 0.f, s2 = 0.f;
#pragma unroll
    for (int i = 0; i < 4; ++i) {
        vals[i] = xr[t + i * 256];
        s += vals[i]; s2 += vals[i] * vals[i];
    }
    __shared__ float rs[256], rq[256];
    rs[t] = s; rq[t] = s2;
    __syncthreads();
    for (int o = 128; o > 0; o >>= 1) {
        if (t < o) { rs[t] += rs[t + o]; rq[t] += rq[t + o]; }
        __syncthreads();
    }
    float mu = rs[0] * (1.f / CEMB);
    float var = rq[0] * (1.f / CEMB) - mu * mu;
    float rstd = rsqrtf(var + 1e-5f);
#pragma unroll
    for (int i = 0; i < 4; ++i) {
        int c = t + i * 256;
        out[(size_t)row * CEMB + c] = (bf16)((vals[i] - mu) * rstd * g[c] + b[c]);
    }
}

// ---------------------------------------------------------------------------
// Tiled WMMA GEMM: C[M,N] = A[M,K](bf16) x W(bf16, stored transposed [N,K]) + bias
// 256 threads = 8 waves. Block tile 128x128, BK=32 (one WMMA K-step).
// Double-buffered LDS filled by async global->LDS copies (ASYNCcnt), so the
// next tiles stream in while the current 8 WMMAs issue.
// ---------------------------------------------------------------------------
#define EPI_QKV  0
#define EPI_RES  1
#define EPI_GELU 2

template <int EPI>
__global__ __launch_bounds__(256)
void gemm_bf16(const bf16* __restrict__ A, const bf16* __restrict__ BT,
               const float* __restrict__ bias, int K, int N,
               const float* __restrict__ res, float* __restrict__ outf,
               bf16* __restrict__ outb,
               bf16* __restrict__ oq, bf16* __restrict__ ok, bf16* __restrict__ ov) {
    __shared__ __align__(16) unsigned short lsA[2][128 * 40];  // 80B rows
    __shared__ __align__(16) unsigned short lsB[2][128 * 40];

    const int tid  = threadIdx.x;
    const int lane = tid & 31, wave = tid >> 5;
    const int wm = wave & 3, wn = wave >> 2;
    const int bm = blockIdx.y * 128, bn = blockIdx.x * 128;
    const int ln15 = lane & 15;
    const int kh   = (lane < 16) ? 0 : 8;    // A-frag K half select
    const int kb   = (lane < 16) ? 0 : 16;   // unused; B-frag loads use kh

    const f32x8 zero = {0.f, 0.f, 0.f, 0.f, 0.f, 0.f, 0.f, 0.f};
    f32x8 acc[2][4];
#pragma unroll
    for (int mt = 0; mt < 2; ++mt)
#pragma unroll
        for (int nt = 0; nt < 4; ++nt) acc[mt][nt] = zero;

    // Async issue of one 128x32 A tile + one 128x32 B tile into buffer `b`.
    auto issue = [&](int b, int kk) {
#pragma unroll
        for (int i = 0; i < 2; ++i) {
            int u = tid + i * 256;
            int r = u >> 2, c = u & 3;
            async_copy_b128(ldsoff(&lsA[b][r * 40 + c * 8]),
                            A + (size_t)(bm + r) * K + kk + c * 8);
            async_copy_b128(ldsoff(&lsB[b][r * 40 + c * 8]),
                            BT + (size_t)(bn + r) * K + kk + c * 8);
        }
    };

    issue(0, 0);
    wait_async0();
    __syncthreads();

    const int nIter = K >> 5;
    for (int it = 0; it < nIter; ++it) {
        const int cur = it & 1;
        if (it + 1 < nIter) issue(cur ^ 1, (it + 1) << 5);  // overlap w/ compute

        AF af[2], bfm[4];
#pragma unroll
        for (int mt = 0; mt < 2; ++mt) {
            int r = wm * 32 + mt * 16 + ln15;
            af[mt].u[0] = *(const U4*)&lsA[cur][r * 40 + kh];
            af[mt].u[1] = *(const U4*)&lsA[cur][r * 40 + 16 + kh];
        }
#pragma unroll
        for (int nt = 0; nt < 4; ++nt) {
            int c = wn * 64 + nt * 16 + ln15;
            bfm[nt].u[0] = *(const U4*)&lsB[cur][c * 40 + kh];
            bfm[nt].u[1] = *(const U4*)&lsB[cur][c * 40 + 16 + kh];
        }
#pragma unroll
        for (int mt = 0; mt < 2; ++mt)
#pragma unroll
            for (int nt = 0; nt < 4; ++nt)
                acc[mt][nt] = wmma_bf16(af[mt], bfm[nt], acc[mt][nt]);

        wait_async0();      // our async fills for the next buffer are complete
        __syncthreads();    // everyone done reading `cur` / writing `cur^1`
    }

    // Epilogue. C/D layout: VGPR r = row r (lanes 0-15) / row r+8 (lanes 16-31).
    const int hi8 = (lane < 16) ? 0 : 8;
#pragma unroll
    for (int mt = 0; mt < 2; ++mt)
#pragma unroll
        for (int nt = 0; nt < 4; ++nt)
#pragma unroll
            for (int r = 0; r < 8; ++r) {
                int row = bm + wm * 32 + mt * 16 + r + hi8;
                int col = bn + wn * 64 + nt * 16 + ln15;
                float vsum = acc[mt][nt][r] + bias[col];
                if (EPI == EPI_QKV) {
                    int b = row >> 10, n = row & 1023;
                    int part = col >> 10, rem = col & 1023;
                    int hh = rem >> 6, dd = rem & 63;
                    bf16* dst = (part == 0) ? oq : (part == 1) ? ok : ov;
                    dst[((size_t)((b << 4) + hh) * 1024 + n) * 64 + dd] = (bf16)vsum;
                } else if (EPI == EPI_RES) {
                    size_t idx = (size_t)row * N + col;
                    outf[idx] = res[idx] + vsum;
                } else { // EPI_GELU: exact erf GELU -> bf16
                    float gv = 0.5f * vsum * (1.f + erff(vsum * 0.70710678118654752f));
                    outb[(size_t)row * N + col] = (bf16)gv;
                }
            }
}

// ---------------------------------------------------------------------------
// Flash attention: q,k,v bf16 [B*H, 1024, 64] -> out bf16 [B,1024,1024]
// Block = 128 threads (4 waves) = one (b,h) x 64 query rows; 16 rows per wave.
// K chunk copied global->LDS with async ops; V transposed through VGPRs.
// ---------------------------------------------------------------------------
__global__ __launch_bounds__(128)
void attn_kernel(const bf16* __restrict__ Q, const bf16* __restrict__ Kb,
                 const bf16* __restrict__ Vb, bf16* __restrict__ out) {
    __shared__ __align__(16) unsigned short kt[32 * 72];   // K chunk [key][d]
    __shared__ __align__(16) unsigned short vt[64 * 40];   // V chunk transposed [d][key]
    __shared__ __align__(16) unsigned short ps[4 * 16 * 40]; // per-wave P staging

    const int tid = threadIdx.x, lane = tid & 31, w = tid >> 5;
    const int bh = blockIdx.x >> 4;      // b*16 + h
    const int qc = blockIdx.x & 15;      // 64-row query chunk
    const int b = bh >> 4, hh = bh & 15;
    const int ln15 = lane & 15;
    const int kh  = (lane < 16) ? 0 : 8;
    const int hi8 = (lane < 16) ? 0 : 8;
    unsigned short* psw = &ps[w * 640];

    const bf16* kg = Kb + (size_t)bh * 1024 * 64;
    const bf16* vg = Vb + (size_t)bh * 1024 * 64;
    const bf16* qp = Q + ((size_t)bh * 1024 + qc * 64 + w * 16) * 64;

    // Q A-fragments for the two K=32 chunks of head_dim=64 (held all kernel)
    AF qf[2];
#pragma unroll
    for (int c = 0; c < 2; ++c) {
        size_t off = (size_t)ln15 * 64 + c * 32 + kh;
        qf[c].u[0] = *(const U4*)(qp + off);
        qf[c].u[1] = *(const U4*)(qp + off + 16);
    }

    const f32x8 zero = {0.f, 0.f, 0.f, 0.f, 0.f, 0.f, 0.f, 0.f};
    f32x8 o[4];
#pragma unroll
    for (int t = 0; t < 4; ++t) o[t] = zero;
    float mst[8], lst[8];
#pragma unroll
    for (int r = 0; r < 8; ++r) { mst[r] = -1e30f; lst[r] = 0.f; }

    for (int j0 = 0; j0 < 1024; j0 += 32) {
        __syncthreads();   // previous iteration done with kt/vt
        // K: 32x64 bf16 = 256 x 16B async copies (2/thread).
        // V: through VGPRs with transpose-on-store into vt[d][key].
#pragma unroll
        for (int i = 0; i < 2; ++i) {
            int u = tid + i * 128;
            int row = u >> 3, colc = u & 7;
            async_copy_b128(ldsoff(&kt[row * 72 + colc * 8]),
                            kg + (size_t)(j0 + row) * 64 + colc * 8);
            union { U4 d; unsigned short s[8]; } e;
            e.d = *(const U4*)(vg + (size_t)(j0 + row) * 64 + colc * 8);
#pragma unroll
            for (int jj = 0; jj < 8; ++jj)
                vt[(colc * 8 + jj) * 40 + row] = e.s[jj];
        }
        wait_async0();
        __syncthreads();

        // S = Q * K^T  (16 x 32 scores as two 16x16 tiles), scaled by 1/sqrt(64)
        f32x8 sc[2];
#pragma unroll
        for (int t = 0; t < 2; ++t) {
            sc[t] = zero;
#pragma unroll
            for (int c = 0; c < 2; ++c) {
                AF kf;
                int key = t * 16 + ln15;
                int ko  = c * 32 + kh;
                kf.u[0] = *(const U4*)&kt[key * 72 + ko];
                kf.u[1] = *(const U4*)&kt[key * 72 + ko + 16];
                sc[t] = wmma_bf16(qf[c], kf, sc[t]);
            }
            sc[t] *= 0.125f;
        }

        // Online softmax per row (row r in vgpr r; lanes 0-15 / 16-31 halves)
#pragma unroll
        for (int r = 0; r < 8; ++r) {
            float mr = fmaxf(sc[0][r], sc[1][r]);
#pragma unroll
            for (int m = 1; m < 16; m <<= 1) mr = fmaxf(mr, __shfl_xor(mr, m, 16));
            float mn = fmaxf(mst[r], mr);
            float al = __expf(mst[r] - mn);
            mst[r] = mn;
            float p0 = __expf(sc[0][r] - mn);
            float p1 = __expf(sc[1][r] - mn);
            sc[0][r] = p0; sc[1][r] = p1;
            float rsum = p0 + p1;
#pragma unroll
            for (int m = 1; m < 16; m <<= 1) rsum += __shfl_xor(rsum, m, 16);
            lst[r] = lst[r] * al + rsum;
#pragma unroll
            for (int t = 0; t < 4; ++t) o[t][r] *= al;
        }

        // Restage P (C-layout) -> A-layout via per-wave LDS scratch
#pragma unroll
        for (int t = 0; t < 2; ++t)
#pragma unroll
            for (int r = 0; r < 8; ++r) {
                HC cv; cv.h = (bf16)sc[t][r];
                psw[(r + hi8) * 40 + t * 16 + ln15] = cv.s;
            }
        asm volatile("s_wait_dscnt 0x0" ::: "memory");  // intra-wave LDS RAW

        AF pf;
        pf.u[0] = *(const U4*)&psw[ln15 * 40 + kh];
        pf.u[1] = *(const U4*)&psw[ln15 * 40 + 16 + kh];

        // O += P * V   (K=32 keys, N=64 head-dim as 4 tiles)
#pragma unroll
        for (int t = 0; t < 4; ++t) {
            AF vf;
            int d  = t * 16 + ln15;
            int ko = kh;
            vf.u[0] = *(const U4*)&vt[d * 40 + ko];
            vf.u[1] = *(const U4*)&vt[d * 40 + ko + 16];
            o[t] = wmma_bf16(pf, vf, o[t]);
        }
    }

    // Normalize and scatter to [B, N, C] (col = h*64 + d) as bf16
#pragma unroll
    for (int r = 0; r < 8; ++r) {
        float inv = 1.f / lst[r];
        int n = qc * 64 + w * 16 + r + hi8;
#pragma unroll
        for (int t = 0; t < 4; ++t) {
            int col = hh * 64 + t * 16 + ln15;
            out[((size_t)b * 1024 + n) * 1024 + col] = (bf16)(o[t][r] * inv);
        }
    }
}

// ---------------------------------------------------------------------------
// Launcher
// ---------------------------------------------------------------------------
extern "C" void kernel_launch(void* const* d_in, const int* in_sizes, int n_in,
                              void* d_out, int out_size, void* d_ws, size_t ws_size,
                              hipStream_t stream) {
    const float* x      = (const float*)d_in[0];
    const float* qkv_w  = (const float*)d_in[1];
    const float* qkv_b  = (const float*)d_in[2];
    const float* proj_w = (const float*)d_in[3];
    const float* proj_b = (const float*)d_in[4];
    const float* fc1_w  = (const float*)d_in[5];
    const float* fc1_b  = (const float*)d_in[6];
    const float* fc2_w  = (const float*)d_in[7];
    const float* fc2_b  = (const float*)d_in[8];
    const float* ln1_g  = (const float*)d_in[9];
    const float* ln1_b  = (const float*)d_in[10];
    const float* ln2_g  = (const float*)d_in[11];
    const float* ln2_b  = (const float*)d_in[12];
    float* out = (float*)d_out;

    char* ws = (char*)d_ws;
    size_t off = 0;
    bf16* qkvWT  = (bf16*)(ws + off); off += (size_t)3072 * 1024 * 2;
    bf16* projWT = (bf16*)(ws + off); off += (size_t)1024 * 1024 * 2;
    bf16* fc1WT  = (bf16*)(ws + off); off += (size_t)4096 * 1024 * 2;
    bf16* fc2WT  = (bf16*)(ws + off); off += (size_t)1024 * 4096 * 2;
    bf16* h_ln   = (bf16*)(ws + off); off += (size_t)MROWS * 1024 * 2;
    size_t qkvRegion = off;            // q/k/v/attn: 4 x 8388608 bf16 elems
    bf16* qb   = (bf16*)(ws + off); off += (size_t)8388608 * 2;
    bf16* kb   = (bf16*)(ws + off); off += (size_t)8388608 * 2;
    bf16* vb   = (bf16*)(ws + off); off += (size_t)8388608 * 2;
    bf16* attn = (bf16*)(ws + off); off += (size_t)8388608 * 2;
    float* x1  = (float*)(ws + off); off += (size_t)MROWS * 1024 * 4;
    bf16* h_mlp = (bf16*)(ws + qkvRegion);   // reuse q/k/v/attn region (67 MB)

    dim3 tb(32, 8);
    wt_transpose<<<dim3(3072 / 32, 1024 / 32), tb, 0, stream>>>(qkv_w, qkvWT, 1024, 3072);
    wt_transpose<<<dim3(1024 / 32, 1024 / 32), tb, 0, stream>>>(proj_w, projWT, 1024, 1024);
    wt_transpose<<<dim3(4096 / 32, 1024 / 32), tb, 0, stream>>>(fc1_w, fc1WT, 1024, 4096);
    wt_transpose<<<dim3(1024 / 32, 4096 / 32), tb, 0, stream>>>(fc2_w, fc2WT, 4096, 1024);

    // ln1 -> h
    ln_kernel<<<MROWS, 256, 0, stream>>>(x, ln1_g, ln1_b, h_ln);
    // qkv = h @ qkv_w + b  -> scatter to q,k,v [B*H,N,64]
    gemm_bf16<EPI_QKV><<<dim3(3072 / 128, MROWS / 128), 256, 0, stream>>>(
        h_ln, qkvWT, qkv_b, 1024, 3072, nullptr, nullptr, nullptr, qb, kb, vb);
    // attention -> attn [B,N,C] bf16
    attn_kernel<<<128 * 16, 128, 0, stream>>>(qb, kb, vb, attn);
    // x1 = x + attn @ proj_w + b
    gemm_bf16<EPI_RES><<<dim3(1024 / 128, MROWS / 128), 256, 0, stream>>>(
        attn, projWT, proj_b, 1024, 1024, x, x1, nullptr, nullptr, nullptr, nullptr);
    // ln2 -> h
    ln_kernel<<<MROWS, 256, 0, stream>>>(x1, ln2_g, ln2_b, h_ln);
    // h_mlp = gelu(h @ fc1_w + b)
    gemm_bf16<EPI_GELU><<<dim3(4096 / 128, MROWS / 128), 256, 0, stream>>>(
        h_ln, fc1WT, fc1_b, 1024, 4096, nullptr, nullptr, h_mlp, nullptr, nullptr, nullptr);
    // out = x1 + h_mlp @ fc2_w + b
    gemm_bf16<EPI_RES><<<dim3(1024 / 128, MROWS / 128), 256, 0, stream>>>(
        h_mlp, fc2WT, fc2_b, 4096, 1024, x1, out, nullptr, nullptr, nullptr, nullptr);
    (void)in_sizes; (void)n_in; (void)out_size; (void)ws_size;
}
